// SAGE_23270132810409
// MI455X (gfx1250) — matrix operand order repaired and checked by reading, not compiled, not run on hardware
//
#include <hip/hip_runtime.h>
#include <math.h>
#include <stdint.h>

// ---------------------------------------------------------------------------
// GraphSAGE (2-layer) for MI455X / gfx1250, wave32.
//   - features kept in f16 end-to-end (x converted once); WMMA f16 -> f32
//   - scatter-mean uses packed-f16 global atomics (global_atomic_pk_add_f16)
//   - SAGE(x) = mean_agg(x) @ Wl^T + b + x @ Wr^T == [agg|x] @ [Wl;Wr]^T + b
//     done as ONE K=192 WMMA GEMM per layer, mean folded into A-staging.
// ---------------------------------------------------------------------------

typedef __attribute__((ext_vector_type(16))) _Float16 v16h;
typedef __attribute__((ext_vector_type(2)))  _Float16 v2h;
typedef __attribute__((ext_vector_type(8)))  float    v8f;

#define NF    96     // feature width (both layers' inputs)
#define KTOT  192    // concatenated K ([agg | self])
#define MBLK  64     // node rows per workgroup
#define TPB   256    // 8 waves of 32
#define EPB   8      // edges per scatter block

union Frag16 { v16h v; unsigned u[8]; };

__device__ __forceinline__ _Float16 f2h(float x) { return (_Float16)x; }

// packed f16 atomic add (GLOBAL_ATOMIC_PK_ADD_F16, no-return -> STOREcnt)
__device__ __forceinline__ void atomic_pk_add_f16(_Float16* addr, v2h val) {
#if __has_builtin(__builtin_amdgcn_global_atomic_fadd_v2f16)
    (void)__builtin_amdgcn_global_atomic_fadd_v2f16(
        (__attribute__((address_space(1))) v2h*)(v2h*)addr, val);
#else
    union { v2h h; unsigned u; } cv; cv.h = val;
    asm volatile("global_atomic_pk_add_f16 %0, %1, off"
                 :: "v"((unsigned long long)(uintptr_t)addr), "v"(cv.u)
                 : "memory");
#endif
}

// ---------------- zero scratch (agg + cnt are bit-zero) ----------------
__global__ void zero_u32(unsigned* __restrict__ p, long long n) {
    long long i = (long long)blockIdx.x * blockDim.x + threadIdx.x;
    long long stride = (long long)gridDim.x * blockDim.x;
    for (; i < n; i += stride) p[i] = 0u;
}

// ---------------- one-time f32 -> f16 conversion of x ----------------
__global__ void cvt_f16(const float* __restrict__ in, _Float16* __restrict__ o,
                        long long n) {
    long long i = (long long)blockIdx.x * blockDim.x + threadIdx.x;
    long long stride = (long long)gridDim.x * blockDim.x;
    for (; i < n; i += stride) o[i] = f2h(in[i]);
}

// ---------------- in-degree counts ----------------
__global__ void degree_k(const int* __restrict__ ei, int E, float* __restrict__ cnt) {
    int e = blockIdx.x * blockDim.x + threadIdx.x;
    if (e < E) atomicAdd(&cnt[ei[E + e]], 1.0f);
}

// ---------------- scatter-add of f16 feature rows ----------------
// blockDim.x == 96: lanes split as (pair = tid%48, sub = tid/48);
// 48 lanes cover one edge's 96 features as 48 packed-f16 atomics,
// two edges in flight per block iteration.
__global__ void scatter_h(const _Float16* __restrict__ feat, const int* __restrict__ ei,
                          int E, _Float16* __restrict__ agg) {
    __shared__ int ss[EPB], sd[EPB];
    const int tid  = threadIdx.x;
    const int base = blockIdx.x * EPB;
    if (tid < EPB) {
        int e = base + tid;
        ss[tid] = (e < E) ? ei[e]     : -1;
        sd[tid] = (e < E) ? ei[E + e] : -1;
    }
    __syncthreads();
    const int p   = (tid % 48) * 2;   // feature index (even)
    const int sub = tid / 48;         // 0 or 1
    for (int i = sub; i < EPB; i += 2) {
        int s = ss[i];
        if (s < 0) continue;
        if (i + 2 < EPB && ss[i + 2] >= 0)
            __builtin_prefetch(&feat[(size_t)ss[i + 2] * NF + p], 0, 0);
        v2h val = *(const v2h*)&feat[(size_t)s * NF + p];
        atomic_pk_add_f16(&agg[(size_t)sd[i] * NF + p], val);
    }
}

// ---------------- fused SAGE linear: [agg_mean | self] @ [Wl;Wr]^T + b ----
// One block = 64 nodes. A (64x192 f16) and B^T (96x192 f16) staged in LDS.
// Each wave computes 16x16 output tiles with v_wmma_f32_16x16x32_f16.
template <typename TOUT>
__global__ __launch_bounds__(TPB)
void sage_wmma(const _Float16* __restrict__ aggh, const float* __restrict__ cnt,
               const _Float16* __restrict__ self, const float* __restrict__ Wl,
               const float* __restrict__ bias, const float* __restrict__ Wr,
               TOUT* __restrict__ out, int nnodes, int nout, int do_relu) {
    __shared__ __align__(16) _Float16 lsA[MBLK * KTOT];  // [row][k]
    __shared__ __align__(16) _Float16 lsB[NF * KTOT];    // [outchan][k] (B^T)

    const int tid   = threadIdx.x;
    const int gbase = blockIdx.x * MBLK;

    // Stage weights: lsB[n][k] = (k<96 ? Wl[n][k] : Wr[n][k-96]); zero-pad n>=nout.
    for (int i = tid; i < NF * KTOT; i += TPB) {
        int n = i / KTOT, k = i - n * KTOT;
        float v = 0.0f;
        if (n < nout) v = (k < NF) ? Wl[n * NF + k] : Wr[n * NF + (k - NF)];
        lsB[i] = f2h(v);
    }
    // Stage activations; mean normalization folded in; self path is a raw f16 copy.
    for (int i = tid; i < MBLK * KTOT; i += TPB) {
        int r = i / KTOT, k = i - r * KTOT;
        int g = gbase + r;
        _Float16 v = (_Float16)0.0f;
        if (g < nnodes) {
            if (k < NF) {
                float inv = 1.0f / fmaxf(cnt[g], 1.0f);
                v = f2h((float)aggh[(size_t)g * NF + k] * inv);
            } else {
                v = self[(size_t)g * NF + (k - NF)];
            }
        }
        lsA[i] = v;
    }
    __syncthreads();

    const int lane = tid & 31;
    const int wave = tid >> 5;
    const int half = lane >> 4;
    const int l16  = lane & 15;
    const int ntc    = (nout + 15) >> 4;     // 16-wide col tiles (6 or 3)
    const int ntiles = ntc * (MBLK / 16);

    for (int t = wave; t < ntiles; t += (TPB / 32)) {
        const int mr = (t / ntc) * 16;
        const int nc = (t % ntc) * 16;
        const int arow = mr + l16;
        const int bcol = nc + l16;
        v8f c = {0.f, 0.f, 0.f, 0.f, 0.f, 0.f, 0.f, 0.f};

        #pragma unroll
        for (int kb = 0; kb < KTOT; kb += 32) {
            Frag16 a, b;
            // A 16x32 f16 layout (ISA 7.12.2): lanes0-15 K=0..7/16..23, lanes16-31 +8
            #pragma unroll
            for (int j = 0; j < 4; ++j) {
                int k0 = kb + half * 8 + 2 * j;
                a.u[j]     = *(const unsigned*)&lsA[arow * KTOT + k0];
                a.u[4 + j] = *(const unsigned*)&lsA[arow * KTOT + k0 + 16];
            }
            // B 32x16 f16 layout: lanes0-15 K=0..7/16..23, lanes16-31 +8 (2/VGPR)
            #pragma unroll
            for (int j = 0; j < 4; ++j) {
                int k0 = kb + half * 8 + 2 * j;
                b.u[j]     = *(const unsigned*)&lsB[bcol * KTOT + k0];
                b.u[4 + j] = *(const unsigned*)&lsB[bcol * KTOT + k0 + 16];
            }
            c = __builtin_amdgcn_wmma_f32_16x16x32_f16(
                    false, a.v, false, b.v, (short)0, c, false, false);
        }

        // C/D layout: VGPR r -> M = r + 8*half, N = lane%16
        if (bcol < nout) {
            float bv = bias[bcol];
            #pragma unroll
            for (int r = 0; r < 8; ++r) {
                int grow = gbase + mr + half * 8 + r;
                if (grow < nnodes) {
                    float v = c[r] + bv;
                    if (do_relu) v = fmaxf(v, 0.0f);
                    out[(size_t)grow * nout + bcol] = (TOUT)v;
                }
            }
        }
    }
}

// ---------------- row-wise log_softmax over 40 classes ----------------
__global__ void logsoftmax_k(float* __restrict__ out, int nnodes, int ncls) {
    int g = blockIdx.x * blockDim.x + threadIdx.x;
    if (g >= nnodes) return;
    float* row = out + (size_t)g * ncls;
    float m = -3.0e38f;
    for (int j = 0; j < ncls; ++j) m = fmaxf(m, row[j]);
    float s = 0.0f;
    for (int j = 0; j < ncls; ++j) s += expf(row[j] - m);
    float lse = m + logf(s);
    for (int j = 0; j < ncls; ++j) row[j] = row[j] - lse;
}

// ---------------------------------------------------------------------------
extern "C" void kernel_launch(void* const* d_in, const int* in_sizes, int n_in,
                              void* d_out, int out_size, void* d_ws, size_t ws_size,
                              hipStream_t stream) {
    const float* x   = (const float*)d_in[0];
    const int*   ei  = (const int*)  d_in[1];   // [2, E] edge index
    const float* W1l = (const float*)d_in[2];
    const float* b1  = (const float*)d_in[3];
    const float* W1r = (const float*)d_in[4];
    const float* W2l = (const float*)d_in[5];
    const float* b2  = (const float*)d_in[6];
    const float* W2r = (const float*)d_in[7];
    float* out = (float*)d_out;

    const int N = in_sizes[0] / NF;     // 50000
    const int E = in_sizes[1] / 2;      // 800000
    const int NCLS = 40;

    // workspace (bytes): xh | h | agg | cnt   (agg,cnt contiguous for one zero pass)
    char* ws = (char*)d_ws;
    const size_t S = (size_t)N * NF * sizeof(_Float16);
    _Float16* xh  = (_Float16*)(ws);
    _Float16* h   = (_Float16*)(ws + S);
    _Float16* agg = (_Float16*)(ws + 2 * S);
    float*    cnt = (float*)   (ws + 3 * S);

    const long long aggWords = (long long)N * (NF / 2);   // agg as u32 words

    // ---- prep ----
    cvt_f16<<<1024, 256, 0, stream>>>(x, xh, (long long)N * NF);
    zero_u32<<<1024, 256, 0, stream>>>((unsigned*)agg, aggWords + N);  // agg + cnt
    degree_k<<<(E + 255) / 256, 256, 0, stream>>>(ei, E, cnt);

    // ---- layer 1 ----
    scatter_h<<<(E + EPB - 1) / EPB, NF, 0, stream>>>(xh, ei, E, agg);
    sage_wmma<_Float16><<<(N + MBLK - 1) / MBLK, TPB, 0, stream>>>(
        agg, cnt, xh, W1l, b1, W1r, h, N, NF, /*relu=*/1);

    // ---- layer 2 ----
    zero_u32<<<1024, 256, 0, stream>>>((unsigned*)agg, aggWords);
    scatter_h<<<(E + EPB - 1) / EPB, NF, 0, stream>>>(h, ei, E, agg);
    sage_wmma<float><<<(N + MBLK - 1) / MBLK, TPB, 0, stream>>>(
        agg, cnt, h, W2l, b2, W2r, out, N, NCLS, /*relu=*/0);

    logsoftmax_k<<<(N + 255) / 256, 256, 0, stream>>>(out, N, NCLS);
}
